// GptOssExpertsLora_6691559047538
// MI455X (gfx1250) — matrix operand order repaired and checked by reading, not compiled, not run on hardware
//
#include <hip/hip_runtime.h>
#include <math.h>

// Problem constants (from reference setup_inputs)
#define T_TOK   1024
#define H_DIM   1024
#define I_DIM   2048
#define E_NUM   8
#define K_TOP   2
#define R_RANK  16
#define TWOI    4096
#define NPAIR   (T_TOK * K_TOP)          // 2048 routed (token,expert) pairs
#define LIMIT_F 7.0f
#define ALPHA_F 1.702f
#define SCALE_F 2.0f

// GEMM tiling: 256 threads = 8 wave32. 4x4 grid of 16x16 WMMA tiles -> 64x64
// block tile; each wave owns 2 M-subtiles x 1 N-subtile (2 accumulators) and
// each staging round covers K=64 (two WMMA K-steps) -> 4 WMMAs per barrier.
#define BM   64
#define BN   64
#define BK   64
#define BKP  (BK + 4)                    // padded LDS stride for B^T tile
#define MBLK ((NPAIR + BM - 1) / BM)     // 32 max row-blocks per expert
#define A2_LD 64                         // lora 'a' row stride (zero-padded)

typedef __bf16 bf16_t;
typedef bf16_t       v16bf __attribute__((ext_vector_type(16)));
typedef float        v8f   __attribute__((ext_vector_type(8)));
typedef unsigned int v8u   __attribute__((ext_vector_type(8)));

__device__ __forceinline__ unsigned short f32_to_bf16(float f) {
    unsigned int u = __float_as_uint(f);
    u += 0x7fffu + ((u >> 16) & 1u);          // round-to-nearest-even
    return (unsigned short)(u >> 16);
}
__device__ __forceinline__ float bf16_to_f32(unsigned short h) {
    return __uint_as_float(((unsigned int)h) << 16);
}
__device__ __forceinline__ unsigned pack_bf16x2(float lo, float hi) {
    return (unsigned)f32_to_bf16(lo) | ((unsigned)f32_to_bf16(hi) << 16);
}

// CDNA5 async global->LDS copy: 16B per lane, tracked by ASYNCcnt.
// vdst VGPR carries the per-lane LDS byte address (low 32 bits of the generic
// shared pointer), vaddr the 64-bit global address.
__device__ __forceinline__ void async_ld_b128(unsigned lds_off, const void* g) {
    unsigned long long ga = (unsigned long long)g;
    asm volatile("global_load_async_to_lds_b128 %0, %1, off"
                 :: "v"(lds_off), "v"(ga) : "memory");
}
__device__ __forceinline__ void wait_async0() {
    asm volatile("s_wait_asynccnt 0x0" ::: "memory");
}
__device__ __forceinline__ unsigned lds_off_of(const void* p) {
    return (unsigned)(unsigned long long)p;   // LDS offset = low 32 bits
}

// One 16x16x32 bf16 WMMA. Abase = &As[row*BK + ks*32]; Bbase = &BsT[n*BKP + ks*32].
// Fragment packing per CDNA5 ISA 7.12.2 (A: K pairs across 8 VGPRs, halves of the
// wave split K by 8; B^T: same K order along the padded LDS row, halves split K by 8).
__device__ __forceinline__ v8f wmma_frag(const unsigned short* Abase,
                                         const unsigned short* Bbase,
                                         int half, v8f c) {
    v8u au, bu;
#pragma unroll
    for (int v = 0; v < 8; ++v) {
        int ka = (v & 3) * 2 + (v >> 2) * 16 + half * 8;
        au[v] = *(const unsigned int*)(Abase + ka);
        bu[v] = *(const unsigned int*)(Bbase + ka);
    }
    return __builtin_amdgcn_wmma_f32_16x16x32_bf16(
        false, __builtin_bit_cast(v16bf, au),
        false, __builtin_bit_cast(v16bf, bu),
        (short)0, c, false, false);
}

// ---------------------------------------------------------------- utilities
// f32 -> bf16, 8 elements per thread, 128-bit loads/stores
__global__ void convert_x_kernel(const float* __restrict__ x,
                                 unsigned short* __restrict__ xb) {
    int i = blockIdx.x * blockDim.x + threadIdx.x;       // 0 .. T*H/8-1
    const float4 a = ((const float4*)x)[2 * i];
    const float4 b = ((const float4*)x)[2 * i + 1];
    uint4 o;
    o.x = pack_bf16x2(a.x, a.y);
    o.y = pack_bf16x2(a.z, a.w);
    o.z = pack_bf16x2(b.x, b.y);
    o.w = pack_bf16x2(b.z, b.w);
    ((uint4*)xb)[i] = o;
}

// Bin the 2048 pairs by expert (single block). Slot order inside an expert is
// atomic-order dependent, but each pair's row arithmetic is independent of its
// slot and the combine gathers via inv_slot -> bitwise-deterministic output.
__global__ void bin_kernel(const int* __restrict__ rid,
                           const float* __restrict__ rw,
                           int* __restrict__ counts, int* __restrict__ offsets,
                           int* __restrict__ pair_token, int* __restrict__ pair_expert,
                           float* __restrict__ pair_w, int* __restrict__ inv_slot) {
    __shared__ int cnt_s[E_NUM], off_s[E_NUM], cur_s[E_NUM];
    int tid = threadIdx.x;
    if (tid < E_NUM) { cnt_s[tid] = 0; cur_s[tid] = 0; }
    __syncthreads();
    for (int p = tid; p < NPAIR; p += blockDim.x)
        atomicAdd(&cnt_s[rid[p]], 1);
    __syncthreads();
    if (tid == 0) {
        int acc = 0;
        for (int e = 0; e < E_NUM; ++e) { off_s[e] = acc; acc += cnt_s[e]; }
    }
    __syncthreads();
    for (int p = tid; p < NPAIR; p += blockDim.x) {
        int e = rid[p];
        int t = p >> 1;                      // p = t*K_TOP + k
        int slot = off_s[e] + atomicAdd(&cur_s[e], 1);
        pair_token[slot]  = t;
        pair_expert[slot] = e;
        pair_w[slot]      = rw[(size_t)t * E_NUM + e];
        inv_slot[p]       = slot;
    }
    __syncthreads();
    if (tid < E_NUM) { counts[tid] = cnt_s[tid]; offsets[tid] = off_s[tid]; }
}

// a2gu[slot][0..15] = 2 * (x[t] @ lora_A_gate_up[e]) (bf16, zero-padded to 64)
__global__ __launch_bounds__(128) void lora_a_gu_kernel(
    const float* __restrict__ x, const float* __restrict__ Agu,
    const int* __restrict__ pair_token, const int* __restrict__ pair_expert,
    unsigned short* __restrict__ a2gu) {
    const int s = blockIdx.x;
    const int e = pair_expert[s];
    const int t = pair_token[s];
    const int tid = threadIdx.x;
    const int r = tid & 15, hg = tid >> 4;   // 16 ranks x 8 h-groups
    __shared__ float part[8][16];
    float acc = 0.f;
    const int h0 = hg * (H_DIM / 8);
    for (int h = h0; h < h0 + H_DIM / 8; ++h)
        acc += x[(size_t)t * H_DIM + h] * Agu[((size_t)e * H_DIM + h) * R_RANK + r];
    part[hg][r] = acc;
    __syncthreads();
    if (tid < 16) {
        float v = 0.f;
        for (int g = 0; g < 8; ++g) v += part[g][tid];
        a2gu[(size_t)s * A2_LD + tid] = f32_to_bf16(SCALE_F * v);
    } else if (tid < A2_LD) {
        a2gu[(size_t)s * A2_LD + tid] = 0;
    }
}

// a2dn[slot][0..15] = 2 * (gated[slot] @ lora_A_down[e])
__global__ __launch_bounds__(128) void lora_a_dn_kernel(
    const unsigned short* __restrict__ gated, const float* __restrict__ Adn,
    const int* __restrict__ pair_expert, unsigned short* __restrict__ a2dn) {
    const int s = blockIdx.x;
    const int e = pair_expert[s];
    const int tid = threadIdx.x;
    const int r = tid & 15, hg = tid >> 4;
    __shared__ float part[8][16];
    float acc = 0.f;
    const int i0 = hg * (I_DIM / 8);
    for (int i = i0; i < i0 + I_DIM / 8; ++i)
        acc += bf16_to_f32(gated[(size_t)s * I_DIM + i]) *
               Adn[((size_t)e * I_DIM + i) * R_RANK + r];
    part[hg][r] = acc;
    __syncthreads();
    if (tid < 16) {
        float v = 0.f;
        for (int g = 0; g < 8; ++g) v += part[g][tid];
        a2dn[(size_t)s * A2_LD + tid] = f32_to_bf16(SCALE_F * v);
    } else if (tid < A2_LD) {
        a2dn[(size_t)s * A2_LD + tid] = 0;
    }
}

// ---------------------------------------------------------------- stage 1
// gu = gather(x) @ W_gu[e] + bias + (2*x@A) @ B ; activation ; gated -> ws (bf16)
__global__ __launch_bounds__(256) void gemm_gateup_kernel(
    const unsigned short* __restrict__ x_bf, const float* __restrict__ Wgu,
    const float* __restrict__ bias_gu, const unsigned short* __restrict__ a2gu,
    const float* __restrict__ Bgu, const int* __restrict__ counts,
    const int* __restrict__ offsets, const int* __restrict__ pair_token,
    unsigned short* __restrict__ gated) {
    const int e   = blockIdx.y / MBLK;
    const int mb  = blockIdx.y % MBLK;
    const int cnt = counts[e];
    const int m0  = mb * BM;
    if (m0 >= cnt) return;
    const int soff = offsets[e];
    const int n0   = blockIdx.x * BN;

    __shared__ __align__(16) unsigned short As[BM * BK];
    __shared__ __align__(16) unsigned short BsT[BN * BKP];
    __shared__ float Cs[BM][BN];
    __shared__ int toks[BM];

    const int tid = threadIdx.x;
    if (tid < BM) {
        int mm = m0 + tid;
        toks[tid] = (mm < cnt) ? pair_token[soff + mm] : 0;
    }
    __syncthreads();

    const int lane = tid & 31, wave = tid >> 5;
    const int wn = wave & 3, wmg = wave >> 2;         // N-subtile / M-pair
    const int half = lane >> 4, ml = lane & 15;
    const int fr0 = wmg * 32 + ml, fr1 = fr0 + 16;

    // per-thread fixed A-staging chunks (16B each): 2 chunks cover 64x64 bf16
    const int c0 = tid, c1 = tid + 256;
    const int r0 = c0 >> 3, o0 = (c0 & 7) * 8;
    const int r1 = c1 >> 3, o1 = (c1 & 7) * 8;
    const unsigned ldsA0 = lds_off_of(&As[r0 * BK + o0]);
    const unsigned ldsA1 = lds_off_of(&As[r1 * BK + o1]);
    const unsigned short* gA0 = x_bf + (size_t)toks[r0] * H_DIM + o0;
    const unsigned short* gA1 = x_bf + (size_t)toks[r1] * H_DIM + o1;
    const float* Wp = Wgu + (size_t)e * H_DIM * TWOI + n0;

    v8f c0a = {0.f,0.f,0.f,0.f,0.f,0.f,0.f,0.f};
    v8f c1a = c0a;

    for (int k0 = 0; k0 < H_DIM; k0 += BK) {
        async_ld_b128(ldsA0, gA0 + k0);               // A tile via async path
        async_ld_b128(ldsA1, gA1 + k0);
#pragma unroll
        for (int it = 0; it < 8; ++it) {              // B tile: f32 -> bf16^T
            int i = tid + it * 256;                   // 0..2047
            int kk2 = i >> 6, nn = i & 63;
            float f0 = Wp[(size_t)(k0 + 2 * kk2) * TWOI + nn];
            float f1 = Wp[(size_t)(k0 + 2 * kk2 + 1) * TWOI + nn];
            *(unsigned*)&BsT[nn * BKP + 2 * kk2] = pack_bf16x2(f0, f1);
        }
        wait_async0();
        __syncthreads();
#pragma unroll
        for (int ks = 0; ks < 2; ++ks) {
            const unsigned short* Bb = BsT + (wn * 16 + ml) * BKP + ks * 32;
            c0a = wmma_frag(As + fr0 * BK + ks * 32, Bb, half, c0a);
            c1a = wmma_frag(As + fr1 * BK + ks * 32, Bb, half, c1a);
        }
        __syncthreads();
    }

    // LoRA correction as one extra round (A pre-scaled by 2, K padded 16->64)
    {
        int ra0 = min(soff + m0 + r0, NPAIR - 1);
        int ra1 = min(soff + m0 + r1, NPAIR - 1);
        async_ld_b128(ldsA0, a2gu + (size_t)ra0 * A2_LD + o0);
        async_ld_b128(ldsA1, a2gu + (size_t)ra1 * A2_LD + o1);
#pragma unroll
        for (int it = 0; it < 8; ++it) {
            int i = tid + it * 256;
            int kk2 = i >> 6, nn = i & 63;
            float f0 = (2 * kk2     < R_RANK) ? Bgu[((size_t)e * R_RANK + 2 * kk2)     * TWOI + n0 + nn] : 0.f;
            float f1 = (2 * kk2 + 1 < R_RANK) ? Bgu[((size_t)e * R_RANK + 2 * kk2 + 1) * TWOI + n0 + nn] : 0.f;
            *(unsigned*)&BsT[nn * BKP + 2 * kk2] = pack_bf16x2(f0, f1);
        }
        wait_async0();
        __syncthreads();
#pragma unroll
        for (int ks = 0; ks < 2; ++ks) {
            const unsigned short* Bb = BsT + (wn * 16 + ml) * BKP + ks * 32;
            c0a = wmma_frag(As + fr0 * BK + ks * 32, Bb, half, c0a);
            c1a = wmma_frag(As + fr1 * BK + ks * 32, Bb, half, c1a);
        }
        __syncthreads();
    }

#pragma unroll
    for (int r8 = 0; r8 < 8; ++r8) {
        Cs[wmg * 32      + half * 8 + r8][wn * 16 + ml] = c0a[r8];
        Cs[wmg * 32 + 16 + half * 8 + r8][wn * 16 + ml] = c1a[r8];
    }
    __syncthreads();

    // bias + clamp/SiLU gating on (even, odd) column pairs; 64x32 gated values
#pragma unroll
    for (int it = 0; it < 8; ++it) {
        int i = tid + it * 256;                       // 0..2047
        int m = i >> 5, j = i & 31;
        if (m0 + m < cnt) {
            float g = Cs[m][2 * j]     + bias_gu[(size_t)e * TWOI + n0 + 2 * j];
            float u = Cs[m][2 * j + 1] + bias_gu[(size_t)e * TWOI + n0 + 2 * j + 1];
            g = fminf(g, LIMIT_F);
            u = fminf(fmaxf(u, -LIMIT_F), LIMIT_F);
            float act = (u + 1.0f) * (g / (1.0f + __expf(-ALPHA_F * g)));
            gated[(size_t)(soff + m0 + m) * I_DIM + (n0 >> 1) + j] = f32_to_bf16(act);
        }
    }
}

// ---------------------------------------------------------------- stage 2
// out = gated @ W_dn[e] + bias + (2*gated@A) @ B ; scale by routing weight
__global__ __launch_bounds__(256) void gemm_down_kernel(
    const unsigned short* __restrict__ gated, const float* __restrict__ Wdn,
    const float* __restrict__ bias_dn, const unsigned short* __restrict__ a2dn,
    const float* __restrict__ Bdn, const int* __restrict__ counts,
    const int* __restrict__ offsets, const float* __restrict__ pair_w,
    float* __restrict__ out_ws) {
    const int e   = blockIdx.y / MBLK;
    const int mb  = blockIdx.y % MBLK;
    const int cnt = counts[e];
    const int m0  = mb * BM;
    if (m0 >= cnt) return;
    const int soff = offsets[e];
    const int n0   = blockIdx.x * BN;

    __shared__ __align__(16) unsigned short As[BM * BK];
    __shared__ __align__(16) unsigned short BsT[BN * BKP];
    __shared__ float Cs[BM][BN];

    const int tid = threadIdx.x;
    const int lane = tid & 31, wave = tid >> 5;
    const int wn = wave & 3, wmg = wave >> 2;
    const int half = lane >> 4, ml = lane & 15;
    const int fr0 = wmg * 32 + ml, fr1 = fr0 + 16;

    const int c0 = tid, c1 = tid + 256;
    const int r0 = c0 >> 3, o0 = (c0 & 7) * 8;
    const int r1 = c1 >> 3, o1 = (c1 & 7) * 8;
    const unsigned ldsA0 = lds_off_of(&As[r0 * BK + o0]);
    const unsigned ldsA1 = lds_off_of(&As[r1 * BK + o1]);
    const int ra0 = min(soff + m0 + r0, NPAIR - 1);   // clamped: invalid rows are
    const int ra1 = min(soff + m0 + r1, NPAIR - 1);   // masked at the epilogue
    const unsigned short* gA0 = gated + (size_t)ra0 * I_DIM + o0;
    const unsigned short* gA1 = gated + (size_t)ra1 * I_DIM + o1;
    const float* Wp = Wdn + (size_t)e * I_DIM * H_DIM + n0;

    v8f c0a = {0.f,0.f,0.f,0.f,0.f,0.f,0.f,0.f};
    v8f c1a = c0a;

    for (int k0 = 0; k0 < I_DIM; k0 += BK) {
        async_ld_b128(ldsA0, gA0 + k0);
        async_ld_b128(ldsA1, gA1 + k0);
#pragma unroll
        for (int it = 0; it < 8; ++it) {
            int i = tid + it * 256;
            int kk2 = i >> 6, nn = i & 63;
            float f0 = Wp[(size_t)(k0 + 2 * kk2) * H_DIM + nn];
            float f1 = Wp[(size_t)(k0 + 2 * kk2 + 1) * H_DIM + nn];
            *(unsigned*)&BsT[nn * BKP + 2 * kk2] = pack_bf16x2(f0, f1);
        }
        wait_async0();
        __syncthreads();
#pragma unroll
        for (int ks = 0; ks < 2; ++ks) {
            const unsigned short* Bb = BsT + (wn * 16 + ml) * BKP + ks * 32;
            c0a = wmma_frag(As + fr0 * BK + ks * 32, Bb, half, c0a);
            c1a = wmma_frag(As + fr1 * BK + ks * 32, Bb, half, c1a);
        }
        __syncthreads();
    }

    // LoRA round
    {
        async_ld_b128(ldsA0, a2dn + (size_t)ra0 * A2_LD + o0);
        async_ld_b128(ldsA1, a2dn + (size_t)ra1 * A2_LD + o1);
#pragma unroll
        for (int it = 0; it < 8; ++it) {
            int i = tid + it * 256;
            int kk2 = i >> 6, nn = i & 63;
            float f0 = (2 * kk2     < R_RANK) ? Bdn[((size_t)e * R_RANK + 2 * kk2)     * H_DIM + n0 + nn] : 0.f;
            float f1 = (2 * kk2 + 1 < R_RANK) ? Bdn[((size_t)e * R_RANK + 2 * kk2 + 1) * H_DIM + n0 + nn] : 0.f;
            *(unsigned*)&BsT[nn * BKP + 2 * kk2] = pack_bf16x2(f0, f1);
        }
        wait_async0();
        __syncthreads();
#pragma unroll
        for (int ks = 0; ks < 2; ++ks) {
            const unsigned short* Bb = BsT + (wn * 16 + ml) * BKP + ks * 32;
            c0a = wmma_frag(As + fr0 * BK + ks * 32, Bb, half, c0a);
            c1a = wmma_frag(As + fr1 * BK + ks * 32, Bb, half, c1a);
        }
        __syncthreads();
    }

#pragma unroll
    for (int r8 = 0; r8 < 8; ++r8) {
        Cs[wmg * 32      + half * 8 + r8][wn * 16 + ml] = c0a[r8];
        Cs[wmg * 32 + 16 + half * 8 + r8][wn * 16 + ml] = c1a[r8];
    }
    __syncthreads();

    // epilogue: bias + routing-weight scale, 128-bit stores
#pragma unroll
    for (int it = 0; it < 4; ++it) {
        int idx = tid + it * 256;                     // 0..1023
        int m = idx >> 4, q = idx & 15;
        if (m0 + m < cnt) {
            float w = pair_w[soff + m0 + m];
            const float4 b = *(const float4*)&bias_dn[(size_t)e * H_DIM + n0 + q * 4];
            float4 o;
            o.x = (Cs[m][q * 4 + 0] + b.x) * w;
            o.y = (Cs[m][q * 4 + 1] + b.y) * w;
            o.z = (Cs[m][q * 4 + 2] + b.z) * w;
            o.w = (Cs[m][q * 4 + 3] + b.w) * w;
            *(float4*)&out_ws[(size_t)(soff + m0 + m) * H_DIM + n0 + q * 4] = o;
        }
    }
}

// y[t] = out_ws[slot(t,0)] + out_ws[slot(t,1)]   (fully overwrites d_out)
__global__ void combine_kernel(const float* __restrict__ out_ws,
                               const int* __restrict__ inv_slot,
                               float* __restrict__ y) {
    int t = blockIdx.x;
    int s0 = inv_slot[2 * t], s1 = inv_slot[2 * t + 1];
    int h4 = threadIdx.x;                             // 256 float4 per row
    const float4 a = *(const float4*)&out_ws[(size_t)s0 * H_DIM + h4 * 4];
    const float4 b = *(const float4*)&out_ws[(size_t)s1 * H_DIM + h4 * 4];
    float4 o = {a.x + b.x, a.y + b.y, a.z + b.z, a.w + b.w};
    *(float4*)&y[(size_t)t * H_DIM + h4 * 4] = o;
}

// ---------------------------------------------------------------- launch
extern "C" void kernel_launch(void* const* d_in, const int* in_sizes, int n_in,
                              void* d_out, int out_size, void* d_ws, size_t ws_size,
                              hipStream_t stream) {
    const float* x   = (const float*)d_in[0];
    const int*   rid = (const int*)d_in[1];
    const float* rw  = (const float*)d_in[2];
    const float* Wgu = (const float*)d_in[3];
    const float* bgu = (const float*)d_in[4];
    const float* Wdn = (const float*)d_in[5];
    const float* bdn = (const float*)d_in[6];
    const float* Agu = (const float*)d_in[7];
    const float* Bgu = (const float*)d_in[8];
    const float* Adn = (const float*)d_in[9];
    const float* Bdn = (const float*)d_in[10];
    float* y = (float*)d_out;
    (void)in_sizes; (void)n_in; (void)out_size; (void)ws_size;

    // workspace carve-up (~19.5 MB total)
    char* ws = (char*)d_ws;
    size_t off = 0;
    auto take = [&](size_t bytes) -> char* {
        char* p = ws + off;
        off += (bytes + 255) & ~(size_t)255;
        return p;
    };
    unsigned short* x_bf   = (unsigned short*)take((size_t)T_TOK * H_DIM * 2);
    unsigned short* gated  = (unsigned short*)take((size_t)NPAIR * I_DIM * 2);
    float*          out_ws = (float*)take((size_t)NPAIR * H_DIM * 4);
    unsigned short* a2gu   = (unsigned short*)take((size_t)NPAIR * A2_LD * 2);
    unsigned short* a2dn   = (unsigned short*)take((size_t)NPAIR * A2_LD * 2);
    int*   pair_token  = (int*)take(NPAIR * 4);
    int*   pair_expert = (int*)take(NPAIR * 4);
    float* pair_w      = (float*)take(NPAIR * 4);
    int*   inv_slot    = (int*)take(NPAIR * 4);
    int*   counts      = (int*)take(E_NUM * 4);
    int*   offsets     = (int*)take(E_NUM * 4);

    hipLaunchKernelGGL(convert_x_kernel, dim3((T_TOK * H_DIM / 8) / 256), dim3(256),
                       0, stream, x, x_bf);
    hipLaunchKernelGGL(bin_kernel, dim3(1), dim3(256), 0, stream, rid, rw, counts,
                       offsets, pair_token, pair_expert, pair_w, inv_slot);
    hipLaunchKernelGGL(lora_a_gu_kernel, dim3(NPAIR), dim3(128), 0, stream, x, Agu,
                       pair_token, pair_expert, a2gu);
    hipLaunchKernelGGL(gemm_gateup_kernel, dim3(TWOI / BN, E_NUM * MBLK), dim3(256),
                       0, stream, x_bf, Wgu, bgu, a2gu, Bgu, counts, offsets,
                       pair_token, gated);
    hipLaunchKernelGGL(lora_a_dn_kernel, dim3(NPAIR), dim3(128), 0, stream, gated,
                       Adn, pair_expert, a2dn);
    hipLaunchKernelGGL(gemm_down_kernel, dim3(H_DIM / BN, E_NUM * MBLK), dim3(256),
                       0, stream, gated, Wdn, bdn, a2dn, Bdn, counts, offsets,
                       pair_w, out_ws);
    hipLaunchKernelGGL(combine_kernel, dim3(T_TOK), dim3(256), 0, stream, out_ws,
                       inv_slot, y);
}
